// GATNetHeadsChanged4LayersLeakyReLU_LayerNorm_50388556317211
// MI455X (gfx1250) — compile-verified, matrix operand order checked by reading, not run on hardware
//
#include <hip/hip_runtime.h>
#include <hip/hip_bf16.h>

typedef __attribute__((ext_vector_type(16))) __bf16 v16bf;
typedef __attribute__((ext_vector_type(8)))  __bf16 v8bf;
typedef __attribute__((ext_vector_type(8)))  float  v8f;
typedef __attribute__((ext_vector_type(4)))  float  v4f;

__device__ __forceinline__ float lrelu(float v, float s){ return v >= 0.f ? v : s * v; }

// Monotonic float <-> uint encoding so segment-max can use a single HW atomicMax(u32).
__device__ __forceinline__ unsigned fenc(float f){
  unsigned u = __float_as_uint(f);
  return ((int)u >= 0) ? (u | 0x80000000u) : ~u;
}
__device__ __forceinline__ float fdec(unsigned k){
  return (k & 0x80000000u) ? __uint_as_float(k & 0x7FFFFFFFu) : __uint_as_float(~k);
}

// ---------------- utility kernels ----------------
__global__ void k_fill_u32(unsigned* __restrict__ p, unsigned v, int n){
  int i = blockIdx.x * blockDim.x + threadIdx.x;
  if (i < n) p[i] = v;
}

__global__ void k_cvt_bf16(const float* __restrict__ s, __bf16* __restrict__ d, int n){
  int i = blockIdx.x * blockDim.x + threadIdx.x;
  if (i < n) d[i] = (__bf16)s[i];
}

// ---------------- WMMA GEMM: C[M,Nc] = A[M,K] * B[Nc,K]^T + bias ----------------
// A,B bf16 row-major; C f32. Requires M%16==0, Nc%16==0, K%32==0.
// One wave per 16x16 C tile, 8 waves (128 cols) per block along N.
__global__ void k_gemm_bf16(const __bf16* __restrict__ A, const __bf16* __restrict__ B,
                            const float* __restrict__ bias, float* __restrict__ C,
                            int M, int Nc, int K){
  int wave = threadIdx.x >> 5;
  int lane = threadIdx.x & 31;
  int m0 = blockIdx.x << 4;
  int n0 = (blockIdx.y * 8 + wave) << 4;
  if (n0 >= Nc || m0 >= M) return;          // wave-uniform: EXEC stays all-ones for WMMA
  int l15 = lane & 15;
  int hi  = lane >> 4;                      // 0: K {0..7,16..23}; 1: K {8..15,24..31}
  const __bf16* ap = A + (size_t)(m0 + l15) * K + hi * 8;
  const __bf16* bp = B + (size_t)(n0 + l15) * K + hi * 8;
  v8f c = {0.f,0.f,0.f,0.f,0.f,0.f,0.f,0.f};
  for (int k0 = 0; k0 < K; k0 += 32){
    v16bf a, b;
    ((v8bf*)&a)[0] = *(const v8bf*)(ap + k0);
    ((v8bf*)&a)[1] = *(const v8bf*)(ap + k0 + 16);
    ((v8bf*)&b)[0] = *(const v8bf*)(bp + k0);
    ((v8bf*)&b)[1] = *(const v8bf*)(bp + k0 + 16);
    c = __builtin_amdgcn_wmma_f32_16x16x32_bf16(false, a, false, b, (short)0, c, false, false);
  }
  int col = n0 + l15;
  float bc = bias ? bias[col] : 0.f;
  #pragma unroll
  for (int r = 0; r < 8; ++r){
    int row = m0 + r + hi * 8;              // C layout: VGPR r -> M=r (lanes 0-15) / M=8+r (lanes 16-31)
    C[(size_t)row * Nc + col] = c[r] + bc;
  }
}

// ---------------- GAT attention coefficients ----------------
__global__ void k_att(const float* __restrict__ hgat, const float* __restrict__ att_s,
                      const float* __restrict__ att_d, float* __restrict__ as_,
                      float* __restrict__ ad_, int N){
  int t = blockIdx.x * blockDim.x + threadIdx.x;
  if (t >= N * 2) return;
  int n = t >> 1, h = t & 1;
  const float* hp = hgat + (size_t)n * 256 + h * 128;
  const float* ws = att_s + h * 128;
  const float* wd = att_d + h * 128;
  float ss = 0.f, sd = 0.f;
  for (int f = 0; f < 128; ++f){ float v = hp[f]; ss += v * ws[f]; sd += v * wd[f]; }
  as_[t] = ss; ad_[t] = sd;
}

// ---------------- edge softmax passes (E real edges + N self-loops) ----------------
__global__ void k_edge_logits(const int* __restrict__ ei, const float* __restrict__ as_,
                              const float* __restrict__ ad_, float* __restrict__ ebuf,
                              unsigned* __restrict__ mkey, int E, int N){
  int idx = blockIdx.x * blockDim.x + threadIdx.x;
  if (idx >= E + N) return;
  int s, d;
  if (idx < E){ s = ei[idx]; d = ei[E + idx]; } else { s = d = idx - E; }
  #pragma unroll
  for (int h = 0; h < 2; ++h){
    float e = lrelu(as_[s * 2 + h] + ad_[d * 2 + h], 0.2f);
    ebuf[(size_t)idx * 2 + h] = e;
    atomicMax(&mkey[d * 2 + h], fenc(e));
  }
}

__global__ void k_edge_exp(const int* __restrict__ ei, float* __restrict__ ebuf,
                           const unsigned* __restrict__ mkey, float* __restrict__ ssum,
                           int E, int N){
  int idx = blockIdx.x * blockDim.x + threadIdx.x;
  if (idx >= E + N) return;
  int d = (idx < E) ? ei[E + idx] : idx - E;
  #pragma unroll
  for (int h = 0; h < 2; ++h){
    float ex = expf(ebuf[(size_t)idx * 2 + h] - fdec(mkey[d * 2 + h]));
    ebuf[(size_t)idx * 2 + h] = ex;
    atomicAdd(&ssum[d * 2 + h], ex);
  }
}

// 64 threads per edge, each owns 4 contiguous features (float4 load, 4 f32 atomics).
__global__ void k_edge_agg(const int* __restrict__ ei, const float* __restrict__ ebuf,
                           const float* __restrict__ ssum, const float* __restrict__ hgat,
                           float* __restrict__ agg, int E, int N){
  long long t = (long long)blockIdx.x * blockDim.x + threadIdx.x;
  if (t >= (long long)(E + N) * 64) return;
  int idx = (int)(t >> 6);
  int q   = (int)(t & 63);      // feature quad: features [q*4, q*4+3]
  int h   = q >> 5;             // head = feature/128
  int s, d;
  if (idx < E){ s = ei[idx]; d = ei[E + idx]; } else { s = d = idx - E; }
  float alpha = ebuf[(size_t)idx * 2 + h] / (ssum[d * 2 + h] + 1e-16f);
  v4f hv = *(const v4f*)(hgat + (size_t)s * 256 + q * 4);
  float* op = agg + (size_t)d * 256 + q * 4;
  atomicAdd(op + 0, hv.x * alpha);
  atomicAdd(op + 1, hv.y * alpha);
  atomicAdd(op + 2, hv.z * alpha);
  atomicAdd(op + 3, hv.w * alpha);
}

// agg + b_gat, lrelu(0.01), convert to bf16 for next WMMA GEMM
__global__ void k_bias_lrelu_bf16(const float* __restrict__ in, const float* __restrict__ bias,
                                  __bf16* __restrict__ out, int n, int F){
  int i = blockIdx.x * blockDim.x + threadIdx.x;
  if (i >= n) return;
  float v = lrelu(in[i] + bias[i % F], 0.01f);
  out[i] = (__bf16)v;
}

// ---------------- LayerNorm + lrelu, one wave32 per row ----------------
__global__ void k_ln_lrelu(const float* __restrict__ in, const float* __restrict__ w,
                           const float* __restrict__ b, float* __restrict__ outf,
                           __bf16* __restrict__ outb, int rows, int F){
  int wave = threadIdx.x >> 5;
  int lane = threadIdx.x & 31;
  int row = blockIdx.x * 8 + wave;
  if (row >= rows) return;
  const float* rp = in + (size_t)row * F;
  float vv[4];
  float sum = 0.f, sumsq = 0.f;
  int cnt = 0;
  for (int f = lane; f < F; f += 32){ float v = rp[f]; vv[cnt++] = v; sum += v; sumsq += v * v; }
  #pragma unroll
  for (int off = 16; off > 0; off >>= 1){
    sum   += __shfl_xor(sum,   off, 32);
    sumsq += __shfl_xor(sumsq, off, 32);
  }
  float mean = sum / F;
  float var  = sumsq / F - mean * mean;
  float rstd = rsqrtf(var + 1e-5f);
  cnt = 0;
  for (int f = lane; f < F; f += 32){
    float y = lrelu((vv[cnt++] - mean) * rstd * w[f] + b[f], 0.01f);
    if (outf) outf[(size_t)row * F + f] = y;
    if (outb) outb[(size_t)row * F + f] = (__bf16)y;
  }
}

// ---------------- final 32->3 projection + packed (x,y,z,|.|^2) ----------------
__global__ void k_head3(const float* __restrict__ h, const float* __restrict__ W3,
                        const float* __restrict__ b3, v4f* __restrict__ p, int N){
  int i = blockIdx.x * blockDim.x + threadIdx.x;
  if (i >= N) return;
  const float* v = h + (size_t)i * 32;
  float y0 = b3[0], y1 = b3[1], y2 = b3[2];
  #pragma unroll
  for (int k = 0; k < 32; ++k){
    float x = v[k];
    y0 += W3[k] * x; y1 += W3[32 + k] * x; y2 += W3[64 + k] * x;
  }
  v4f r; r.x = y0; r.y = y1; r.z = y2; r.w = y0*y0 + y1*y1 + y2*y2;
  p[i] = r;
}

// ---------------- pairwise distances: 400 MB of NT float4 stores ----------------
__global__ void k_cdist(const v4f* __restrict__ p, float* __restrict__ out, int N){
  int i = blockIdx.y * 16 + threadIdx.y;
  int j = (blockIdx.x * 16 + threadIdx.x) * 4;
  if (i >= N || j >= N) return;               // N%4==0 so whole float4 stays in range
  v4f pi = p[i];
  v4f r;
  #pragma unroll
  for (int k = 0; k < 4; ++k){
    v4f q = p[j + k];
    float d2 = pi.w + q.w - 2.f * (pi.x*q.x + pi.y*q.y + pi.z*q.z);
    d2 = fmaxf(d2, 0.f);
    float dd = d2 > 0.f ? sqrtf(d2) : 0.f;
    ((float*)&r)[k] = dd;
  }
  __builtin_nontemporal_store(r, (v4f*)(out + (size_t)i * N + j));
}

extern "C" void kernel_launch(void* const* d_in, const int* in_sizes, int n_in,
                              void* d_out, int out_size, void* d_ws, size_t ws_size,
                              hipStream_t stream){
  const float* x     = (const float*)d_in[0];
  const int*   ei    = (const int*)  d_in[1];
  const float* W_gat = (const float*)d_in[2];
  const float* b_gat = (const float*)d_in[3];
  const float* att_s = (const float*)d_in[4];
  const float* att_d = (const float*)d_in[5];
  const float* Wa    = (const float*)d_in[6];
  const float* ba    = (const float*)d_in[7];
  const float* lna_w = (const float*)d_in[8];
  const float* lna_b = (const float*)d_in[9];
  const float* W1    = (const float*)d_in[10];
  const float* b1    = (const float*)d_in[11];
  const float* ln1_w = (const float*)d_in[12];
  const float* ln1_b = (const float*)d_in[13];
  const float* W2    = (const float*)d_in[14];
  const float* b2    = (const float*)d_in[15];
  const float* ln2_w = (const float*)d_in[16];
  const float* ln2_b = (const float*)d_in[17];
  const float* W3    = (const float*)d_in[18];
  const float* b3    = (const float*)d_in[19];

  const int N = in_sizes[0] / 256;
  const int E = in_sizes[1] / 2;

  // workspace carve-up (256B aligned)
  char* base = (char*)d_ws;
  size_t off = 0;
  auto alloc = [&](size_t bytes)->char*{
    off = (off + 255) & ~(size_t)255;
    char* p = base + off;
    off += bytes;
    return p;
  };
  __bf16* xb  = (__bf16*)alloc((size_t)N * 256 * 2);
  __bf16* Wgb = (__bf16*)alloc(256 * 256 * 2);
  __bf16* Wab = (__bf16*)alloc(128 * 256 * 2);
  __bf16* W1b = (__bf16*)alloc(64 * 128 * 2);
  __bf16* W2b = (__bf16*)alloc(32 * 64 * 2);
  float*  hgat = (float*)alloc((size_t)N * 256 * 4);
  float*  as_  = (float*)alloc((size_t)N * 2 * 4);
  float*  ad_  = (float*)alloc((size_t)N * 2 * 4);
  unsigned* mkey = (unsigned*)alloc((size_t)N * 2 * 4);
  float*  ssum = (float*)alloc((size_t)N * 2 * 4);
  float*  ebuf = (float*)alloc((size_t)(E + N) * 2 * 4);
  float*  agg  = (float*)alloc((size_t)N * 256 * 4);
  __bf16* h0b  = (__bf16*)alloc((size_t)N * 256 * 2);
  float*  ra   = (float*)alloc((size_t)N * 128 * 4);
  __bf16* h1b  = (__bf16*)alloc((size_t)N * 128 * 2);
  float*  r1   = (float*)alloc((size_t)N * 64 * 4);
  __bf16* h2b  = (__bf16*)alloc((size_t)N * 64 * 2);
  float*  r2   = (float*)alloc((size_t)N * 32 * 4);
  float*  h3f  = (float*)alloc((size_t)N * 32 * 4);
  v4f*    p    = (v4f*)alloc((size_t)N * 16);

  const int T = 256;
  auto nb = [](long long n, int t){ return (unsigned)((n + t - 1) / t); };

  // init accumulators (mkey = enc(-inf) = 0x007FFFFF)
  k_fill_u32<<<nb(N*2, T), T, 0, stream>>>(mkey, 0x007FFFFFu, N * 2);
  k_fill_u32<<<nb(N*2, T), T, 0, stream>>>((unsigned*)ssum, 0u, N * 2);
  k_fill_u32<<<nb((long long)N*256, T), T, 0, stream>>>((unsigned*)agg, 0u, N * 256);

  // fp32 -> bf16 for WMMA operands
  k_cvt_bf16<<<nb((long long)N*256, T), T, 0, stream>>>(x, xb, N * 256);
  k_cvt_bf16<<<nb(256*256, T), T, 0, stream>>>(W_gat, Wgb, 256 * 256);
  k_cvt_bf16<<<nb(128*256, T), T, 0, stream>>>(Wa, Wab, 128 * 256);
  k_cvt_bf16<<<nb(64*128, T), T, 0, stream>>>(W1, W1b, 64 * 128);
  k_cvt_bf16<<<nb(32*64, T), T, 0, stream>>>(W2, W2b, 32 * 64);

  // GAT linear: hgat = x @ W_gat^T + b_gat
  k_gemm_bf16<<<dim3(N/16, (256 + 127)/128), T, 0, stream>>>(xb, Wgb, b_gat, hgat, N, 256, 256);

  // attention logits + edge softmax + aggregation
  k_att<<<nb((long long)N*2, T), T, 0, stream>>>(hgat, att_s, att_d, as_, ad_, N);
  k_edge_logits<<<nb((long long)(E+N), T), T, 0, stream>>>(ei, as_, ad_, ebuf, mkey, E, N);
  k_edge_exp<<<nb((long long)(E+N), T), T, 0, stream>>>(ei, ebuf, mkey, ssum, E, N);
  k_edge_agg<<<nb((long long)(E+N)*64, T), T, 0, stream>>>(ei, ebuf, ssum, hgat, agg, E, N);

  // lrelu + bf16 for MLP entry
  k_bias_lrelu_bf16<<<nb((long long)N*256, T), T, 0, stream>>>(agg, b_gat, h0b, N * 256, 256);

  // MLP with WMMA GEMMs + LN/lrelu
  k_gemm_bf16<<<dim3(N/16, 1), T, 0, stream>>>(h0b, Wab, ba, ra, N, 128, 256);
  k_ln_lrelu<<<nb(N, 8), T, 0, stream>>>(ra, lna_w, lna_b, nullptr, h1b, N, 128);
  k_gemm_bf16<<<dim3(N/16, 1), T, 0, stream>>>(h1b, W1b, b1, r1, N, 64, 128);
  k_ln_lrelu<<<nb(N, 8), T, 0, stream>>>(r1, ln1_w, ln1_b, nullptr, h2b, N, 64);
  k_gemm_bf16<<<dim3(N/16, 1), T, 0, stream>>>(h2b, W2b, b2, r2, N, 32, 64);
  k_ln_lrelu<<<nb(N, 8), T, 0, stream>>>(r2, ln2_w, ln2_b, h3f, nullptr, N, 32);

  // 32->3 head, pack (x,y,z,sq), then the bandwidth-dominant cdist
  k_head3<<<nb(N, T), T, 0, stream>>>(h3f, W3, b3, p, N);
  k_cdist<<<dim3((N + 63)/64, (N + 15)/16), dim3(16,16), 0, stream>>>(p, (float*)d_out, N);
}
